// LSTM_14173392076948
// MI455X (gfx1250) — matrix operand order repaired and checked by reading, not compiled, not run on hardware
//
#include <hip/hip_runtime.h>
#include <hip/hip_bf16.h>

typedef __attribute__((ext_vector_type(16))) __bf16 v16bf;
typedef __attribute__((ext_vector_type(8)))  float  v8f;
typedef __attribute__((ext_vector_type(4)))  unsigned u32x4;
typedef __attribute__((ext_vector_type(8)))  int      i32x8;
typedef __attribute__((ext_vector_type(4)))  int      i32x4;

#define S_LEN    2048
#define BATCH    64
#define IN_DIM   256
#define HID      512
#define OUT_N    512
#define KTOT     768      // IN_DIM + HID
#define KIT_REC  24       // 768/32
#define KIT_OUT  16       // 512/32
#define N_REC_WG 32
#define N_OUT_WG 4
#define NWG      (N_REC_WG + N_OUT_WG)
#define NTHREADS 256
#define XSTRIDE  264      // 256 + 8 pad (ushorts): 132 dwords/row -> banks advance by 4
#define HSTRIDE  520      // 512 + 8 pad (ushorts): 260 dwords/row -> banks advance by 4
#define ZSTRIDE  520      // 512 + 8 pad (floats)

// ---- workspace layout (bytes) ----
#define OFF_BAR 0u
#define OFF_W4  256u                          // 2048*768*2  = 3,145,728
#define OFF_WO  (OFF_W4 + 3145728u)           // 512*512*2   =   524,288
#define OFF_H0  (OFF_WO + 524288u)            // 64*512*2    =    65,536
#define OFF_H1  (OFF_H0 + 65536u)
#define OFF_C   (OFF_H1 + 65536u)             // 64*512*4    =   131,072

__device__ __forceinline__ unsigned short f2bf(float f) {
  unsigned u = __float_as_uint(f);
  u += 0x7FFFu + ((u >> 16) & 1u);            // round-to-nearest-even
  return (unsigned short)(u >> 16);
}
__device__ __forceinline__ unsigned pack2bf(float a, float b) {
  return (unsigned)f2bf(a) | ((unsigned)f2bf(b) << 16);
}
__device__ __forceinline__ float sigmoidf_(float x) { return 1.0f / (1.0f + __expf(-x)); }

__device__ __forceinline__ v8f wmma_bf16(v16bf a, v16bf b, v8f c) {
  // (neg_a, A, neg_b, B, c_mod, C, reuse_a, reuse_b) -> v_wmma_f32_16x16x32_bf16
  return __builtin_amdgcn_wmma_f32_16x16x32_bf16(false, a, false, b, (short)0, c, false, false);
}

// Tensor Data Mover: DMA h[t-1] tile [64 rows x 512 bf16] global->LDS with
// per-row padding: 256 dwords data (pad_interval=7) then 4 dwords pad
// (pad_amount=3) -> LDS row stride 520 ushorts == HSTRIDE.
__device__ __forceinline__ void tdm_load_h(const unsigned short* gsrc, unsigned lds_off) {
  unsigned long long ga = (unsigned long long)(uintptr_t)gsrc;
  u32x4 g0;
  g0[0] = 1u;                                        // count=1 (valid), no gather
  g0[1] = lds_off;                                   // lds_addr [63:32]
  g0[2] = (unsigned)ga;                              // global_addr lo
  g0[3] = (unsigned)(ga >> 32) | (2u << 30);         // global_addr hi | type=2 ("image")
  i32x8 g1;
  g1[0] = (int)((1u << 16) |                         // data_size = 1 -> 2 bytes
                (1u << 20) |                         // pad_enable
                (7u << 22) |                         // pad_interval = 7 -> 256 dwords
                (3u << 25));                         // pad_amount   = 3 -> 4 dwords
  g1[1] = (int)(512u << 16);                         // tensor_dim0 = 512 (bits 79:48 lo half)
  g1[2] = (int)(64u  << 16);                         // tensor_dim1 = 64  (bits 111:80 lo half)
  g1[3] = (int)(512u << 16);                         // tile_dim0   = 512 (bits 127:112)
  g1[4] = 64;                                        // tile_dim1 = 64, tile_dim2 = 0
  g1[5] = 512;                                       // tensor_dim0_stride = 512 (lo 32)
  g1[6] = 0;                                         // stride hi / dim1_stride lo
  g1[7] = 0;
  i32x4 g2 = {0, 0, 0, 0};
  i32x4 g3 = {0, 0, 0, 0};
  i32x8 g4 = {0, 0, 0, 0, 0, 0, 0, 0};
  __builtin_amdgcn_tensor_load_to_lds(g0, g1, g2, g3, g4, 0);
}

__device__ __forceinline__ void grid_barrier(unsigned* bar) {
  __threadfence();              // make this WG's global writes agent-visible
  __syncthreads();
  if (threadIdx.x == 0) {
    unsigned gen = __hip_atomic_load(&bar[1], __ATOMIC_RELAXED, __HIP_MEMORY_SCOPE_AGENT);
    unsigned arr = __hip_atomic_fetch_add(&bar[0], 1u, __ATOMIC_ACQ_REL, __HIP_MEMORY_SCOPE_AGENT);
    if (arr == (unsigned)(NWG - 1)) {
      __hip_atomic_store(&bar[0], 0u, __ATOMIC_RELAXED, __HIP_MEMORY_SCOPE_AGENT);
      __hip_atomic_fetch_add(&bar[1], 1u, __ATOMIC_RELEASE, __HIP_MEMORY_SCOPE_AGENT);
    } else {
      while (__hip_atomic_load(&bar[1], __ATOMIC_ACQUIRE, __HIP_MEMORY_SCOPE_AGENT) == gen) {
        __builtin_amdgcn_s_sleep(2);
      }
    }
  }
  __syncthreads();
}

// ---- prologue: zero barrier + h buffers + c state (deterministic per launch) ----
__global__ void init_ws(unsigned* bar, unsigned* hc, int nwords) {
  int i = blockIdx.x * blockDim.x + threadIdx.x;
  if (i < 2) bar[i] = 0u;
  for (int j = i; j < nwords; j += gridDim.x * blockDim.x) hc[j] = 0u;
}

// ---- weight swizzle: bf16, WMMA B-fragment order (lane-major, 16 contiguous per lane) ----
// B layout: lane l holds N = l%16, K = kt*32 + (s/8)*16 + (l/16)*8 + s%8, s=0..15
__global__ void swizzleW4(const float* __restrict__ Wi, const float* __restrict__ Wf,
                          const float* __restrict__ Wg, const float* __restrict__ Wo,
                          unsigned short* __restrict__ W4s) {
  int idx = blockIdx.x * 256 + threadIdx.x;   // < 4*32*24*32*16 = 1,572,864
  int s  = idx & 15;
  int l  = (idx >> 4) & 31;
  int rest = idx >> 9;                        // (g*32+ht)*24 + kt
  int kt = rest % 24;
  int gh = rest / 24;
  int ht = gh & 31;
  int g  = gh >> 5;
  int K  = kt * 32 + ((s >> 3) << 4) + ((l >> 4) << 3) + (s & 7);
  int n  = ht * 16 + (l & 15);
  const float* W = (g == 0) ? Wi : (g == 1) ? Wf : (g == 2) ? Wg : Wo;
  W4s[idx] = f2bf(W[n * KTOT + K]);
}

__global__ void swizzleWo(const float* __restrict__ Wh2o, unsigned short* __restrict__ Wos) {
  int idx = blockIdx.x * 256 + threadIdx.x;   // < 32*16*32*16 = 262,144
  int s  = idx & 15;
  int l  = (idx >> 4) & 31;
  int kt = (idx >> 9) & 15;
  int nt = idx >> 13;
  int K  = kt * 32 + ((s >> 3) << 4) + ((l >> 4) << 3) + (s & 7);
  int n  = nt * 16 + (l & 15);
  Wos[idx] = f2bf(Wh2o[n * HID + K]);
}

// ---- persistent fused LSTM kernel ----
__global__ void __launch_bounds__(NTHREADS, 1)
lstm_main(const float* __restrict__ x,
          const float* __restrict__ bi, const float* __restrict__ bff,
          const float* __restrict__ bg, const float* __restrict__ bo,
          const float* __restrict__ bh2o,
          const unsigned short* __restrict__ W4s,
          const unsigned short* __restrict__ Wos,
          unsigned short* __restrict__ hbuf0,
          unsigned short* __restrict__ hbuf1,
          float* __restrict__ cbuf,
          unsigned* __restrict__ bar,
          float* __restrict__ out) {
  // shared arena (recurrent): s_x[64][264] bf16 (33,792B) @0
  //                           s_h[64][520] bf16 (66,560B) @33,792  (TDM destination)
  //                           s_gate 4*64*16 f32 (16,384B) @100,352
  // shared arena (output):    s_z[16][520] f32 (33,280B)  @0       (overlapped)
  __shared__ __align__(16) unsigned char s_arena[116736];
  unsigned short* s_x    = (unsigned short*)s_arena;
  unsigned short* s_h    = (unsigned short*)(s_arena + 33792);
  float*          s_gate = (float*)(s_arena + 100352);
  float*          s_z    = (float*)s_arena;

  const int wg   = blockIdx.x;
  const int tid  = threadIdx.x;
  const int lane = tid & 31;
  const int wave = tid >> 5;

  if (wg < N_REC_WG) {
    // ---- recurrent WG: owns H columns [wg*16, wg*16+16), all 4 gates, all 64 batch rows ----
    const int mtile = wave & 3;            // batch tile (rows mtile*16..+16)
    const int gp    = wave >> 2;           // gate pair: {2gp, 2gp+1}
    const int g0 = 2 * gp, g1 = 2 * gp + 1;
    const int hcol0 = wg * 16;
    const int mrow  = mtile * 16 + (lane & 15);
    const unsigned lds_h_off = (unsigned)(uintptr_t)(const void*)s_h;

    for (int t = 0; t <= S_LEN; ++t) {
      if (t < S_LEN) {
        const unsigned short* hprev = (t & 1) ? hbuf0 : hbuf1;   // h[t-1] lives in buf[(t-1)&1]
        // kick off TDM DMA of h[t-1] -> s_h (row-padded) from wave 0, then overlap
        // with x_t f32->bf16 staging on all waves
        if (wave == 0) tdm_load_h(hprev, lds_h_off);
        const float* xt = x + (size_t)t * (BATCH * IN_DIM);
        for (int i = tid; i < (BATCH * IN_DIM) / 4; i += NTHREADS) {
          const float4 v = ((const float4*)xt)[i];
          const int row = i >> 6;                 // 64 float4 per row of 256
          const int c4  = (i & 63) << 2;
          unsigned* dst = (unsigned*)&s_x[row * XSTRIDE + c4];
          dst[0] = pack2bf(v.x, v.y);
          dst[1] = pack2bf(v.z, v.w);
        }
        if (wave == 0) __builtin_amdgcn_s_wait_tensorcnt(0);
        __syncthreads();

        v8f acc0 = {}; v8f acc1 = {};
        const unsigned short* rowx = &s_x[mrow * XSTRIDE];
        const unsigned short* rowh = &s_h[mrow * HSTRIDE];
        // K iterations 0..7: x region
        for (int kt = 0; kt < 8; ++kt) {
          union { v16bf v; unsigned u[8]; } A;
          const int kb = kt * 32 + ((lane >> 4) << 3);
          #pragma unroll
          for (int r = 0; r < 8; ++r) {
            const int kk = kb + ((r >> 2) << 4) + ((r & 3) << 1);
            A.u[r] = *(const unsigned*)&rowx[kk];
          }
          const v16bf b0 = *(const v16bf*)&W4s[(((g0 * N_REC_WG + wg) * KIT_REC + kt) << 9) + lane * 16];
          const v16bf b1 = *(const v16bf*)&W4s[(((g1 * N_REC_WG + wg) * KIT_REC + kt) << 9) + lane * 16];
          acc0 = wmma_bf16(A.v, b0, acc0);
          acc1 = wmma_bf16(A.v, b1, acc1);
        }
        // K iterations 8..23: h region
        for (int kt = 8; kt < KIT_REC; ++kt) {
          union { v16bf v; unsigned u[8]; } A;
          const int kb = (kt - 8) * 32 + ((lane >> 4) << 3);
          #pragma unroll
          for (int r = 0; r < 8; ++r) {
            const int kk = kb + ((r >> 2) << 4) + ((r & 3) << 1);
            A.u[r] = *(const unsigned*)&rowh[kk];
          }
          const v16bf b0 = *(const v16bf*)&W4s[(((g0 * N_REC_WG + wg) * KIT_REC + kt) << 9) + lane * 16];
          const v16bf b1 = *(const v16bf*)&W4s[(((g1 * N_REC_WG + wg) * KIT_REC + kt) << 9) + lane * 16];
          if (kt + 1 < KIT_REC) {
            __builtin_prefetch(&W4s[(((g0 * N_REC_WG + wg) * KIT_REC + kt + 1) << 9) + lane * 16], 0, 3);
            __builtin_prefetch(&W4s[(((g1 * N_REC_WG + wg) * KIT_REC + kt + 1) << 9) + lane * 16], 0, 3);
          }
          acc0 = wmma_bf16(A.v, b0, acc0);
          acc1 = wmma_bf16(A.v, b1, acc1);
        }

        // stash gate pre-activations: C layout lane l, vgpr r -> m = r + 8*(l/16), n = l%16
        #pragma unroll
        for (int r = 0; r < 8; ++r) {
          const int m = mtile * 16 + r + ((lane >> 4) << 3);
          const int n = lane & 15;
          s_gate[(g0 * BATCH + m) * 16 + n] = acc0[r];
          s_gate[(g1 * BATCH + m) * 16 + n] = acc1[r];
        }
        __syncthreads();

        // fused pointwise: c/h update for owned 64x16 slice
        unsigned short* hnew = (t & 1) ? hbuf1 : hbuf0;
        for (int e = tid; e < BATCH * 16; e += NTHREADS) {
          const int m = e >> 4, n = e & 15;
          const int hc = hcol0 + n;
          const float iv = sigmoidf_(s_gate[(0 * BATCH + m) * 16 + n] + bi[hc]);
          const float fv = sigmoidf_(s_gate[(1 * BATCH + m) * 16 + n] + bff[hc]);
          const float gv = tanhf    (s_gate[(2 * BATCH + m) * 16 + n] + bg[hc]);
          const float ov = sigmoidf_(s_gate[(3 * BATCH + m) * 16 + n] + bo[hc]);
          const float cp = cbuf[m * HID + hc];
          const float cn = fv * cp + iv * gv;
          cbuf[m * HID + hc] = cn;
          hnew[m * HID + hc] = f2bf(ov * tanhf(cn));
        }
        __syncthreads();
      }
      grid_barrier(bar);
    }
  } else {
    // ---- output WG: batch strip [widx*16, widx*16+16); processes step t-1 (pipelined) ----
    const int widx = wg - N_REC_WG;
    const int mrow = widx * 16 + (lane & 15);
    for (int t = 0; t <= S_LEN; ++t) {
      if (t >= 1) {
        const unsigned short* hp = ((t - 1) & 1) ? hbuf1 : hbuf0;
        v8f acc[4] = {};
        for (int kt = 0; kt < KIT_OUT; ++kt) {
          union { v16bf v; unsigned u[8]; } A;
          const int kb = kt * 32 + ((lane >> 4) << 3);
          #pragma unroll
          for (int r = 0; r < 8; ++r) {
            const int kk = kb + ((r >> 2) << 4) + ((r & 3) << 1);
            A.u[r] = *(const unsigned*)&hp[mrow * HID + kk];
          }
          #pragma unroll
          for (int j = 0; j < 4; ++j) {
            const int nt = wave * 4 + j;
            const v16bf bb = *(const v16bf*)&Wos[((nt * KIT_OUT + kt) << 9) + lane * 16];
            acc[j] = wmma_bf16(A.v, bb, acc[j]);
          }
        }
        #pragma unroll
        for (int j = 0; j < 4; ++j) {
          const int nc = (wave * 4 + j) * 16 + (lane & 15);
          #pragma unroll
          for (int r = 0; r < 8; ++r) {
            const int m = r + ((lane >> 4) << 3);
            s_z[m * ZSTRIDE + nc] = acc[j][r] + bh2o[nc];
          }
        }
        __syncthreads();

        // row-wise log-softmax: 16 threads per row
        const int row = tid >> 4;
        const int c0  = tid & 15;
        float mx = -3.4e38f;
        for (int n = c0; n < OUT_N; n += 16) mx = fmaxf(mx, s_z[row * ZSTRIDE + n]);
        #pragma unroll
        for (int off = 8; off >= 1; off >>= 1) mx = fmaxf(mx, __shfl_xor(mx, off, 16));
        float sum = 0.f;
        for (int n = c0; n < OUT_N; n += 16) sum += __expf(s_z[row * ZSTRIDE + n] - mx);
        #pragma unroll
        for (int off = 8; off >= 1; off >>= 1) sum += __shfl_xor(sum, off, 16);
        const float lse = mx + __logf(sum);
        float* orow = out + ((size_t)(t - 1) * BATCH + (widx * 16 + row)) * OUT_N;
        for (int n = c0; n < OUT_N; n += 16) orow[n] = s_z[row * ZSTRIDE + n] - lse;
        __syncthreads();
      }
      grid_barrier(bar);
    }
  }
}

extern "C" void kernel_launch(void* const* d_in, const int* in_sizes, int n_in,
                              void* d_out, int out_size, void* d_ws, size_t ws_size,
                              hipStream_t stream) {
  (void)in_sizes; (void)n_in; (void)out_size; (void)ws_size;
  const float* x    = (const float*)d_in[0];
  const float* Wi   = (const float*)d_in[1];
  const float* bi   = (const float*)d_in[2];
  const float* Wf   = (const float*)d_in[3];
  const float* bf   = (const float*)d_in[4];
  const float* Wg   = (const float*)d_in[5];
  const float* bg   = (const float*)d_in[6];
  const float* Wo   = (const float*)d_in[7];
  const float* bo   = (const float*)d_in[8];
  const float* Wh2o = (const float*)d_in[9];
  const float* bh2o = (const float*)d_in[10];
  float* out = (float*)d_out;

  unsigned char* ws = (unsigned char*)d_ws;
  unsigned*       bar = (unsigned*)(ws + OFF_BAR);
  unsigned short* W4s = (unsigned short*)(ws + OFF_W4);
  unsigned short* Wos = (unsigned short*)(ws + OFF_WO);
  unsigned short* h0  = (unsigned short*)(ws + OFF_H0);
  unsigned short* h1  = (unsigned short*)(ws + OFF_H1);
  float*          cb  = (float*)(ws + OFF_C);

  init_ws<<<256, 256, 0, stream>>>(bar, (unsigned*)(ws + OFF_H0), (65536 + 65536 + 131072) / 4);
  swizzleW4<<<6144, 256, 0, stream>>>(Wi, Wf, Wg, Wo, W4s);
  swizzleWo<<<1024, 256, 0, stream>>>(Wh2o, Wos);
  lstm_main<<<NWG, NTHREADS, 0, stream>>>(x, bi, bf, bg, bo, bh2o, W4s, Wos, h0, h1, cb, bar, out);
}
